// NeuralGraph_77867757076527
// MI455X (gfx1250) — compile-verified, matrix operand order checked by reading, not run on hardware
//
#include <hip/hip_runtime.h>
#include <hip/hip_bf16.h>
#include <math.h>

#define HDIM 32
#define SEQK 9216
#define BN_EPS 1e-5f

typedef __attribute__((ext_vector_type(8)))  float  v8f;
typedef __attribute__((ext_vector_type(16))) __bf16 v16bf;

__device__ __forceinline__ float gelu_exact(float v) {
    return 0.5f * v * (1.0f + erff(v * 0.70710678118654752f));
}
__device__ __forceinline__ void atomAddG(float* p, float v) {
    __hip_atomic_fetch_add(p, v, __ATOMIC_RELAXED, __HIP_MEMORY_SCOPE_AGENT);
}
__device__ __forceinline__ void atomAddL(float* p, float v) {
    __hip_atomic_fetch_add(p, v, __ATOMIC_RELAXED, __HIP_MEMORY_SCOPE_WORKGROUP);
}

// ---------------------------------------------------------------------------
// K1: Y1 = seq @ We   via v_wmma_f32_16x16x32_bf16.
// One wave owns a 16-row x 32-col output tile (two 16x16 accumulators).
// We[k,0:32] is staged transposed into LDS as bf16 in 128-k chunks so each
// lane's B fragment is two contiguous 16B ds loads. A fragments are converted
// f32->bf16 in registers following the ISA 16-bit A-matrix lane layout.
// ---------------------------------------------------------------------------
__global__ void __launch_bounds__(256)
gemm_seq_wmma(const float* __restrict__ seq, const float* __restrict__ We,
              float* __restrict__ Y1, int N) {
    const int lane = threadIdx.x & 31;
    const int wave = threadIdx.x >> 5;
    const int tile = blockIdx.x * 8 + wave;
    const int row0 = tile * 16;
    const bool active = (row0 < N);
    const int half = lane >> 4;    // K-group select
    const int l15  = lane & 15;    // row (A) / col (B,C)

    __shared__ __bf16 sB[HDIM * 128];   // transposed: sB[n*128 + k]

    v8f c0 = {}; v8f c1 = {};
    const float* arow = seq + (size_t)(row0 + l15) * SEQK;

    for (int k0 = 0; k0 < SEQK; k0 += 128) {
        __syncthreads();   // protect previous chunk's reads
        for (int e = threadIdx.x; e < 128 * HDIM; e += 256) {
            int kk = e >> 5, n = e & 31;
            sB[n * 128 + kk] = (__bf16)We[(size_t)(k0 + kk) * HDIM + n];
        }
        __syncthreads();
        if (active) {
            #pragma unroll
            for (int s = 0; s < 4; ++s) {
                const int kb = s * 32;
                v16bf a, b0, b1;
                // A: lanes 0-15 hold K 0..7 (v0-3) and 16..23 (v4-7); lanes 16-31: +8
                const float* p0 = arow + k0 + kb + half * 8;
                #pragma unroll
                for (int j = 0; j < 8; ++j) {
                    a[j]     = (__bf16)p0[j];
                    a[8 + j] = (__bf16)p0[16 + j];
                }
                const __bf16* q0 = &sB[l15 * 128 + kb + half * 8];        // cols 0..15
                const __bf16* q1 = &sB[(l15 + 16) * 128 + kb + half * 8]; // cols 16..31
                #pragma unroll
                for (int j = 0; j < 8; ++j) {
                    b0[j] = q0[j]; b0[8 + j] = q0[16 + j];
                    b1[j] = q1[j]; b1[8 + j] = q1[16 + j];
                }
                c0 = __builtin_amdgcn_wmma_f32_16x16x32_bf16(false, a, false, b0,
                                                             (short)0, c0, false, false);
                c1 = __builtin_amdgcn_wmma_f32_16x16x32_bf16(false, a, false, b1,
                                                             (short)0, c1, false, false);
            }
        }
    }
    if (active) {
        // C/D layout: VGPR r -> M=r (lanes 0-15), M=8+r (lanes 16-31); N=lane&15
        #pragma unroll
        for (int r = 0; r < 8; ++r) {
            int m = row0 + half * 8 + r;
            Y1[(size_t)m * HDIM + l15]      = c0[r];
            Y1[(size_t)m * HDIM + 16 + l15] = c1[r];
        }
    }
}

// ---------------------------------------------------------------------------
// Per-channel sum / sum-of-squares over a [N,32] matrix (for BatchNorm stats).
// Thread's channel is fixed (stride multiple of 32); LDS partials + atomics.
// ---------------------------------------------------------------------------
__global__ void __launch_bounds__(256)
col_stats(const float* __restrict__ M, int total,
          float* __restrict__ sum32, float* __restrict__ sq32) {
    const int t = threadIdx.x;
    const int c = t & 31;
    float s = 0.f, q = 0.f;
    for (int idx = blockIdx.x * 256 + t; idx < total; idx += 256 * gridDim.x) {
        float v = M[idx];
        s += v; q += v * v;
    }
    __shared__ float ls[HDIM], lq[HDIM];
    if (t < HDIM) { ls[t] = 0.f; lq[t] = 0.f; }
    __syncthreads();
    atomAddL(&ls[c], s);
    atomAddL(&lq[c], q);
    __syncthreads();
    if (t < HDIM) {
        atomAddG(&sum32[t], ls[t]);
        atomAddG(&sq32[t],  lq[t]);
    }
}

// sum/sumsq of x and pause -> st[192..195] = {sx, sxx, sp, spp}
__global__ void __launch_bounds__(256)
scalar_stats(const float* __restrict__ x, const float* __restrict__ pause,
             int N, float* __restrict__ st4) {
    float sx = 0.f, sxx = 0.f, sp = 0.f, spp = 0.f;
    for (int i = blockIdx.x * 256 + threadIdx.x; i < N; i += 256 * gridDim.x) {
        float a = x[i], b = pause[i];
        sx += a; sxx += a * a; sp += b; spp += b * b;
    }
    #pragma unroll
    for (int o = 16; o > 0; o >>= 1) {
        sx  += __shfl_xor(sx,  o, 32);
        sxx += __shfl_xor(sxx, o, 32);
        sp  += __shfl_xor(sp,  o, 32);
        spp += __shfl_xor(spp, o, 32);
    }
    if ((threadIdx.x & 31) == 0) {
        atomAddG(&st4[0], sx);  atomAddG(&st4[1], sxx);
        atomAddG(&st4[2], sp);  atomAddG(&st4[3], spp);
    }
}

// ---------------------------------------------------------------------------
// K3: h = gelu(bn(x*Wx)) + gelu(bn(Y1)); hp = gelu(bn(pause*Wp));
//     F = concat(h, hp) @ Wf.   One wave per row, shfl broadcast for matmul.
// Rank-1 branches use scalar stats: mean_n = m*W_n, var_n = v*W_n^2.
// Linear biases feeding BN cancel exactly (train-mode BN) -> skipped.
// ---------------------------------------------------------------------------
__global__ void __launch_bounds__(256)
fuse_fc(const float* __restrict__ x, const float* __restrict__ pause,
        const float* __restrict__ Y1,
        const float* __restrict__ Wx, const float* __restrict__ gx, const float* __restrict__ betax,
        const float* __restrict__ Wp, const float* __restrict__ gp, const float* __restrict__ betap,
        const float* __restrict__ ge, const float* __restrict__ betae,
        const float* __restrict__ Wf,
        const float* __restrict__ st, float* __restrict__ F, int N) {
    const int lane = threadIdx.x & 31;
    const int i = blockIdx.x * 8 + (threadIdx.x >> 5);
    if (i >= N) return;
    const float invN = 1.0f / (float)N;
    const int n = lane;

    const float mx = st[192] * invN, vx = st[193] * invN - mx * mx;
    const float mp = st[194] * invN, vp = st[195] * invN - mp * mp;

    const float wxn = Wx[n];
    float hn = gelu_exact(gx[n] * (x[i] - mx) * wxn * rsqrtf(vx * wxn * wxn + BN_EPS) + betax[n]);

    const float y   = Y1[(size_t)i * HDIM + n];
    const float mY  = st[n] * invN;
    const float vY  = st[32 + n] * invN - mY * mY;
    float hs = gelu_exact(ge[n] * (y - mY) * rsqrtf(vY + BN_EPS) + betae[n]);

    float h = hn + hs;

    const float wpn = Wp[n];
    float hp = gelu_exact(gp[n] * (pause[i] - mp) * wpn * rsqrtf(vp * wpn * wpn + BN_EPS) + betap[n]);

    float acc = 0.f;
    #pragma unroll 8
    for (int k = 0; k < HDIM; ++k) {
        float hk  = __shfl(h,  k, 32);
        float hpk = __shfl(hp, k, 32);
        acc += hk * Wf[k * HDIM + n] + hpk * Wf[(k + HDIM) * HDIM + n];
    }
    F[(size_t)i * HDIM + n] = acc;
}

// K4: H2 = gelu(bn_f(F))   (elementwise)
__global__ void __launch_bounds__(256)
bn_gelu(const float* __restrict__ F, const float* __restrict__ gf,
        const float* __restrict__ betaf, const float* __restrict__ st,
        float* __restrict__ H2, int total, int N) {
    int idx = blockIdx.x * 256 + threadIdx.x;
    if (idx >= total) return;
    int n = idx & 31;
    float invN = 1.0f / (float)N;
    float m = st[64 + n] * invN;
    float v = st[96 + n] * invN - m * m;
    H2[idx] = gelu_exact(gf[n] * (F[idx] - m) * rsqrtf(v + BN_EPS) + betaf[n]);
}

// K5: scatter-sum. One wave = one edge (32 channels): coalesced 128B gather
// of H2[src], native f32 atomic burst into agg[dst].
__global__ void __launch_bounds__(256)
sage_scatter(const int* __restrict__ ei, int E,
             const float* __restrict__ H2, float* __restrict__ agg) {
    int idx = blockIdx.x * 256 + threadIdx.x;
    if (idx >= E * HDIM) return;
    int e = idx >> 5, c = idx & 31;
    int s = ei[e];
    int d = ei[E + e];
    atomAddG(&agg[(size_t)d * HDIM + c], H2[(size_t)s * HDIM + c]);
}

// K6: Zpre = agg @ Wl + H2 @ Wr   (wave per row; bl cancels under BN)
__global__ void __launch_bounds__(256)
sage_linear(const float* __restrict__ agg, const float* __restrict__ H2,
            const float* __restrict__ Wl, const float* __restrict__ Wr,
            float* __restrict__ Zpre, int N) {
    const int lane = threadIdx.x & 31;
    const int i = blockIdx.x * 8 + (threadIdx.x >> 5);
    if (i >= N) return;
    float a  = agg[(size_t)i * HDIM + lane];
    float h2 = H2[(size_t)i * HDIM + lane];
    float acc = 0.f;
    #pragma unroll 8
    for (int k = 0; k < HDIM; ++k) {
        float ak = __shfl(a,  k, 32);
        float hk = __shfl(h2, k, 32);
        acc += ak * Wl[k * HDIM + lane] + hk * Wr[k * HDIM + lane];
    }
    Zpre[(size_t)i * HDIM + lane] = acc;
}

// K7: z = gelu(bn_c(Zpre)); out = relu(z @ Wreg + breg).
// d_out layout: [0,N) = out, [N, N+N*32) = z  (tuple return order).
__global__ void __launch_bounds__(256)
final_head(const float* __restrict__ Zpre, const float* __restrict__ gc,
           const float* __restrict__ betac, const float* __restrict__ Wreg,
           const float* __restrict__ breg, const float* __restrict__ st,
           float* __restrict__ out, float* __restrict__ z, int N) {
    const int lane = threadIdx.x & 31;
    const int i = blockIdx.x * 8 + (threadIdx.x >> 5);
    if (i >= N) return;
    float invN = 1.0f / (float)N;
    float m = st[128 + lane] * invN;
    float v = st[160 + lane] * invN - m * m;
    float zn = gelu_exact(gc[lane] * (Zpre[(size_t)i * HDIM + lane] - m)
                          * rsqrtf(v + BN_EPS) + betac[lane]);
    z[(size_t)i * HDIM + lane] = zn;
    float p = zn * Wreg[lane];
    #pragma unroll
    for (int o = 16; o > 0; o >>= 1) p += __shfl_xor(p, o, 32);
    if (lane == 0) out[i] = fmaxf(p + breg[0], 0.f);
}

// ---------------------------------------------------------------------------
extern "C" void kernel_launch(void* const* d_in, const int* in_sizes, int n_in,
                              void* d_out, int out_size, void* d_ws, size_t ws_size,
                              hipStream_t stream) {
    const float* x     = (const float*)d_in[0];
    const float* seq   = (const float*)d_in[1];
    const float* pause = (const float*)d_in[2];
    const int*   ei    = (const int*)  d_in[3];
    const float* Wx    = (const float*)d_in[4];
    const float* gx    = (const float*)d_in[6];
    const float* betax = (const float*)d_in[7];
    const float* Wp    = (const float*)d_in[8];
    const float* gp    = (const float*)d_in[10];
    const float* betap = (const float*)d_in[11];
    const float* We    = (const float*)d_in[12];
    const float* ge    = (const float*)d_in[14];
    const float* betae = (const float*)d_in[15];
    const float* Wf    = (const float*)d_in[16];
    const float* gf    = (const float*)d_in[18];
    const float* betaf = (const float*)d_in[19];
    const float* Wl    = (const float*)d_in[20];
    const float* Wr    = (const float*)d_in[22];
    const float* gc    = (const float*)d_in[23];
    const float* betac = (const float*)d_in[24];
    const float* Wreg  = (const float*)d_in[25];
    const float* breg  = (const float*)d_in[26];

    const int N  = in_sizes[0];
    const int E  = in_sizes[3] / 2;
    const size_t NH = (size_t)N * HDIM;

    float* ws   = (float*)d_ws;
    float* Y1   = ws;            // [N,32]
    float* F    = ws + NH;       // [N,32]
    float* H2   = ws + 2 * NH;   // [N,32]
    float* agg  = ws + 3 * NH;   // [N,32]
    float* Zpre = ws + 4 * NH;   // [N,32]
    float* st   = ws + 5 * NH;   // 256 floats of stats

    float* outp = (float*)d_out;       // [N]
    float* zout = (float*)d_out + N;   // [N,32]

    hipMemsetAsync(agg, 0, NH * sizeof(float), stream);
    hipMemsetAsync(st,  0, 256 * sizeof(float), stream);

    // 1) big GEMM (bf16 WMMA, HBM-bound)
    {
        int tiles  = (N + 15) / 16;
        int blocks = (tiles + 7) / 8;
        gemm_seq_wmma<<<blocks, 256, 0, stream>>>(seq, We, Y1, N);
    }
    // 2) stats for Y1 and scalar inputs
    col_stats<<<512, 256, 0, stream>>>(Y1, (int)NH, st, st + 32);
    scalar_stats<<<256, 256, 0, stream>>>(x, pause, N, st + 192);

    const int rowBlocks = (N + 7) / 8;
    // 3) fused feature blocks + fc -> F
    fuse_fc<<<rowBlocks, 256, 0, stream>>>(x, pause, Y1, Wx, gx, betax,
                                           Wp, gp, betap, ge, betae, Wf, st, F, N);
    // 4) stats(F) then H2 = gelu(bn(F))
    col_stats<<<512, 256, 0, stream>>>(F, (int)NH, st + 64, st + 96);
    bn_gelu<<<(int)((NH + 255) / 256), 256, 0, stream>>>(F, gf, betaf, st, H2, (int)NH, N);
    // 5) SAGE neighbor sum
    sage_scatter<<<(E * HDIM + 255) / 256, 256, 0, stream>>>(ei, E, H2, agg);
    // 6) Zpre = agg@Wl + H2@Wr
    sage_linear<<<rowBlocks, 256, 0, stream>>>(agg, H2, Wl, Wr, Zpre, N);
    // 7) stats(Zpre), then final BN+GELU+regressor
    col_stats<<<512, 256, 0, stream>>>(Zpre, (int)NH, st + 128, st + 160);
    final_head<<<rowBlocks, 256, 0, stream>>>(Zpre, gc, betac, Wreg, breg, st, outp, zout, N);
}
